// VQEncoder_60052232733035
// MI455X (gfx1250) — matrix operand order repaired and checked by reading, not compiled, not run on hardware
//
#include <hip/hip_runtime.h>
#include <hip/hip_bf16.h>

// ---------- constants (match reference shapes) ----------
#define B_    8
#define CIN_  1024
#define T_    4096
#define L_    2048      // T_/DS
#define D_    1024      // VQ_C
#define NCODE 2048
#define K1_   2048      // CIN_*DS

typedef __attribute__((ext_vector_type(16))) __bf16 v16bf;
typedef __attribute__((ext_vector_type(8)))  __bf16 v8bf;
typedef __attribute__((ext_vector_type(8)))  float  v8f;
typedef __attribute__((ext_vector_type(4)))  int    v4i;

typedef __attribute__((address_space(1))) v4i gas_v4i;
typedef __attribute__((address_space(3))) v4i las_v4i;

// ---------- async copy global->LDS (CDNA5), guarded with sync fallback ----------
#if defined(__has_builtin)
#  if __has_builtin(__builtin_amdgcn_global_load_async_to_lds_b128)
#    define ASYNC_CP 1
#  endif
#endif

static __device__ __forceinline__ void cp16(const unsigned short* g, unsigned short* l) {
#ifdef ASYNC_CP
    __builtin_amdgcn_global_load_async_to_lds_b128(
        (gas_v4i*)(void*)const_cast<unsigned short*>(g),
        (las_v4i*)(void*)l, 0, 0);
#else
    *(v8bf*)l = *(const v8bf*)g;
#endif
}

// wait for ALL of this wave's async copies, then workgroup barrier
static __device__ __forceinline__ void fence_all() {
#ifdef ASYNC_CP
#  if __has_builtin(__builtin_amdgcn_s_wait_asynccnt)
    __builtin_amdgcn_s_wait_asynccnt(0);
#  else
    asm volatile("s_wait_asynccnt 0x0" ::: "memory");
#  endif
#endif
    __syncthreads();
}

// f32 -> bf16 (round to nearest even), stored as u16
static __device__ __forceinline__ unsigned short f2bf(float f) {
    unsigned u = __float_as_uint(f);
    u += 0x7FFFu + ((u >> 16) & 1u);
    return (unsigned short)(u >> 16);
}

static __device__ __forceinline__ v16bf cat8(v8bf lo, v8bf hi) {
    v16bf r;
#pragma unroll
    for (int i = 0; i < 8; ++i) { r[i] = lo[i]; r[i + 8] = hi[i]; }
    return r;
}

// A fragment (16x32 bf16, MxK): lanes 0-15 row=lane K{0..7,16..23}; lanes 16-31 K{8..15,24..31}
static __device__ __forceinline__ v16bf load_frag_a(const unsigned short* Abuf, int lane,
                                                    int rowStride, int kOff) {
    int row = lane & 15;
    int k0  = (lane < 16) ? 0 : 8;
    const unsigned short* p = Abuf + row * rowStride + kOff + k0;
    v8bf lo = *(const v8bf*)p;
    v8bf hi = *(const v8bf*)(p + 16);
    return cat8(lo, hi);
}
// B fragment (32x16 bf16, KxN), LDS tile stored [n][k]
static __device__ __forceinline__ v16bf load_frag_b(const unsigned short* Bbuf, int lane,
                                                    int colTile, int rowStride, int kOff) {
    int col = (colTile << 4) + (lane & 15);
    int k0  = (lane < 16) ? 0 : 8;
    const unsigned short* p = Bbuf + col * rowStride + kOff + k0;
    v8bf lo = *(const v8bf*)p;
    v8bf hi = *(const v8bf*)(p + 16);
    return cat8(lo, hi);
}

static __device__ __forceinline__ v8f wmma_bf16(v16bf a, v16bf b, v8f c) {
    return __builtin_amdgcn_wmma_f32_16x16x32_bf16(false, a, false, b, (short)0, c, false, false);
}

// wave compute: 16x64 strip, KSTEP=64, 8 WMMAs; b-frags in distinct regs to batch ds waits
static __device__ __forceinline__ void compute_tile(const unsigned short* Abase,
                                                    const unsigned short* Bbase,
                                                    int lane, int wm, int wn, v8f acc[4]) {
    const unsigned short* Ab = Abase + wm * 16 * 64;
#pragma unroll
    for (int ks = 0; ks < 64; ks += 32) {
        v16bf a  = load_frag_a(Ab, lane, 64, ks);
        v16bf b0 = load_frag_b(Bbase, lane, wn * 4 + 0, 64, ks);
        v16bf b1 = load_frag_b(Bbase, lane, wn * 4 + 1, 64, ks);
        v16bf b2 = load_frag_b(Bbase, lane, wn * 4 + 2, 64, ks);
        v16bf b3 = load_frag_b(Bbase, lane, wn * 4 + 3, 64, ks);
        acc[0] = wmma_bf16(a, b0, acc[0]);
        acc[1] = wmma_bf16(a, b1, acc[1]);
        acc[2] = wmma_bf16(a, b2, acc[2]);
        acc[3] = wmma_bf16(a, b3, acc[3]);
    }
}

// ---------- f32 -> bf16 bulk conversion (8 elems/thread) ----------
__global__ __launch_bounds__(256) void cvt_bf16(const float* __restrict__ src,
                                                unsigned short* __restrict__ dst, int n8) {
    int i = blockIdx.x * 256 + threadIdx.x;
    if (i >= n8) return;
    const float4* s = (const float4*)src + (size_t)i * 2;
    float4 a = s[0], b = s[1];
    uint4 o;
    o.x = ((unsigned)f2bf(a.y) << 16) | f2bf(a.x);
    o.y = ((unsigned)f2bf(a.w) << 16) | f2bf(a.z);
    o.z = ((unsigned)f2bf(b.y) << 16) | f2bf(b.x);
    o.w = ((unsigned)f2bf(b.w) << 16) | f2bf(b.z);
    ((uint4*)dst)[i] = o;
}

// ---------- prep: half_e2[n] = 0.5*sum_d embed[n,d]^2 ; zero loss ----------
__global__ __launch_bounds__(256) void vq_prep(const float* __restrict__ embed,
                                               float* __restrict__ half_e2,
                                               float* __restrict__ loss) {
    __shared__ float red[256];
    int tid = threadIdx.x;
    int n = blockIdx.x;
    const float4* src = (const float4*)(embed + (size_t)n * D_);
    float4 v = src[tid];
    red[tid] = v.x * v.x + v.y * v.y + v.z * v.z + v.w * v.w;
    __syncthreads();
    for (int off = 128; off > 0; off >>= 1) {
        if (tid < off) red[tid] += red[tid + off];
        __syncthreads();
    }
    if (tid == 0) {
        half_e2[n] = 0.5f * red[0];
        if (n == 0) *loss = 0.0f;
    }
}

// ---------- GEMM1: tokens[b,l,o] ; block tile M=64(l) x N=128(o); waves 4x2 ----------
// double-buffered, async B staging + register-prefetched A (f32->bf16 in flight)
__global__ __launch_bounds__(256) void vq_tokens(const float* __restrict__ x,
                                                 const unsigned short* __restrict__ w_in_bf,
                                                 const float* __restrict__ b_in,
                                                 unsigned short* __restrict__ tokens_bf) {
    __shared__ unsigned short As[2][64 * 64];   // 2 x 8 KB
    __shared__ unsigned short Bs[2][128 * 64];  // 2 x 16 KB
    int tid = threadIdx.x;
    int w = tid >> 5, lane = tid & 31;
    int wm = w >> 1, wn = w & 1;
    int blk = blockIdx.x;
    int o0 = (blk & 7) * 128;          // 8 o-tiles
    int l0 = ((blk >> 3) & 31) * 64;   // 32 l-tiles
    int bb = blk >> 8;                 // batch

    float2 va[8];
    auto stageB = [&](int buf, int kb) {
#pragma unroll
        for (int j = 0; j < 4; ++j) {
            int chunk = tid + j * 256;            // 1024 chunks
            int n = chunk >> 3, kk0 = (chunk & 7) * 8;
            cp16(w_in_bf + (size_t)(o0 + n) * K1_ + kb + kk0, &Bs[buf][n * 64 + kk0]);
        }
    };
    auto gloadA = [&](int kb) {
        int cb = kb >> 1;
#pragma unroll
        for (int pp = 0; pp < 8; ++pp) {
            int p = tid + pp * 256;               // 2048 (c,s)-pairs
            int r = p >> 5, cp = p & 31;
            va[pp] = *(const float2*)(x + ((size_t)bb * CIN_ + (cb + cp)) * T_ + 2 * (l0 + r));
        }
    };
    auto storeA = [&](int buf) {
#pragma unroll
        for (int pp = 0; pp < 8; ++pp) {
            int p = tid + pp * 256;
            int r = p >> 5, cp = p & 31;
            As[buf][r * 64 + 2 * cp]     = f2bf(va[pp].x);
            As[buf][r * 64 + 2 * cp + 1] = f2bf(va[pp].y);
        }
    };

    v8f acc[4] = {};
    stageB(0, 0);
    gloadA(0);
    storeA(0);
    fence_all();
    int cur = 0;
    for (int kb = 0; kb < K1_; kb += 64) {
        int nxt = cur ^ 1;
        bool more = (kb + 64) < K1_;
        if (more) { stageB(nxt, kb + 64); gloadA(kb + 64); }
        compute_tile(As[cur], Bs[cur], lane, wm, wn, acc);
        if (more) storeA(nxt);
        fence_all();
        cur = nxt;
    }

    int rbase = (lane >> 4) << 3;
#pragma unroll
    for (int j = 0; j < 4; ++j) {
        int oc = o0 + wn * 64 + j * 16 + (lane & 15);
        float bias = b_in[oc];
#pragma unroll
        for (int i = 0; i < 8; ++i) {
            int l = l0 + wm * 16 + rbase + i;
            tokens_bf[((size_t)bb * L_ + l) * D_ + oc] = f2bf(acc[j][i] + bias);
        }
    }
}

// ---------- GEMM2 + argmax: block = 64 rows; N scanned in 16 tiles of 128 ----------
__global__ __launch_bounds__(256) void vq_argmin(const unsigned short* __restrict__ tokens_bf,
                                                 const unsigned short* __restrict__ embed_bf,
                                                 const float* __restrict__ half_e2,
                                                 int* __restrict__ idxbuf) {
    __shared__ unsigned short As[2][64 * 64];
    __shared__ unsigned short Bs[2][128 * 64];  // overlaid by 64x128 f32 score tile
    __shared__ float redv[256];
    __shared__ int   redi[256];
    __shared__ float bestv[64];
    __shared__ int   besti[64];
    int tid = threadIdx.x;
    int w = tid >> 5, lane = tid & 31;
    int wm = w >> 1, wn = w & 1;
    int row0 = blockIdx.x * 64;
    if (tid < 64) { bestv[tid] = -3.4e38f; besti[tid] = 0; }

    auto stageA = [&](int buf, int kb) {
#pragma unroll
        for (int j = 0; j < 2; ++j) {
            int chunk = tid + j * 256;            // 512 chunks (64 rows x 8)
            int r = chunk >> 3, kk0 = (chunk & 7) * 8;
            cp16(tokens_bf + (size_t)(row0 + r) * D_ + kb + kk0, &As[buf][r * 64 + kk0]);
        }
    };

    for (int nt = 0; nt < NCODE / 128; ++nt) {
        int n0 = nt * 128;
        auto stageB = [&](int buf, int kb) {
#pragma unroll
            for (int j = 0; j < 4; ++j) {
                int chunk = tid + j * 256;
                int n = chunk >> 3, kk0 = (chunk & 7) * 8;
                cp16(embed_bf + (size_t)(n0 + n) * D_ + kb + kk0, &Bs[buf][n * 64 + kk0]);
            }
        };
        v8f acc[4] = {};
        stageA(0, 0);
        stageB(0, 0);
        fence_all();
        int cur = 0;
        for (int kb = 0; kb < D_; kb += 64) {
            int nxt = cur ^ 1;
            bool more = (kb + 64) < D_;
            if (more) { stageA(nxt, kb + 64); stageB(nxt, kb + 64); }
            compute_tile(As[cur], Bs[cur], lane, wm, wn, acc);
            fence_all();
            cur = nxt;
        }
        // scores -> LDS (overlay both B buffers): score = t.e - 0.5||e||^2
        float* s_tile = (float*)&Bs[0][0];      // 64 x 128 f32 = 32 KB
        int rbase = (lane >> 4) << 3;
#pragma unroll
        for (int j = 0; j < 4; ++j) {
            int cl = wn * 64 + j * 16 + (lane & 15);
            float sub = half_e2[n0 + cl];
#pragma unroll
            for (int i = 0; i < 8; ++i)
                s_tile[(wm * 16 + rbase + i) * 128 + cl] = acc[j][i] - sub;
        }
        __syncthreads();
        {   // 4 threads per row, 32 cols each; tie -> lowest index (argmin-first semantics)
            int r = tid >> 2, c0 = tid & 3;
            float bv = -3.4e38f; int bi = 0x7fffffff;
#pragma unroll 8
            for (int jj = 0; jj < 32; ++jj) {
                int col = c0 + (jj << 2);
                float v = s_tile[r * 128 + col];
                int gi = n0 + col;
                if (v > bv || (v == bv && gi < bi)) { bv = v; bi = gi; }
            }
            redv[tid] = bv; redi[tid] = bi;
        }
        __syncthreads();
        if (tid < 64) {
            float bv = bestv[tid]; int bi = besti[tid];
#pragma unroll
            for (int j = 0; j < 4; ++j) {
                float v = redv[tid * 4 + j]; int gi = redi[tid * 4 + j];
                if (v > bv || (v == bv && gi < bi)) { bv = v; bi = gi; }
            }
            bestv[tid] = bv; besti[tid] = bi;
        }
        __syncthreads();
    }
    if (tid < 64) idxbuf[row0 + tid] = besti[tid];
}

// ---------- commit loss: vectorized (8x uint4 tokens, 16x float4 embed per thread) ----------
__global__ __launch_bounds__(256) void vq_loss(const unsigned short* __restrict__ tokens_bf,
                                               const float* __restrict__ embed,
                                               const int* __restrict__ idxbuf,
                                               float* __restrict__ loss) {
    __shared__ float red[256];
    int tid = threadIdx.x;
    size_t base = ((size_t)blockIdx.x * 256 + tid) * 64;    // 64 elems, within one row
    int row = (int)(base >> 10);
    int d0  = (int)(base & 1023);
    const uint4*  tv4 = (const uint4*)(tokens_bf + base);
    const float4* qv4 = (const float4*)(embed + (size_t)idxbuf[row] * D_ + d0);
    float s = 0.0f;
#pragma unroll
    for (int c = 0; c < 8; ++c) {
        uint4 t = tv4[c];
        float4 q0 = qv4[c * 2], q1 = qv4[c * 2 + 1];
        float d;
        d = q0.x - __uint_as_float(t.x << 16);          s += d * d;
        d = q0.y - __uint_as_float(t.x & 0xFFFF0000u);  s += d * d;
        d = q0.z - __uint_as_float(t.y << 16);          s += d * d;
        d = q0.w - __uint_as_float(t.y & 0xFFFF0000u);  s += d * d;
        d = q1.x - __uint_as_float(t.z << 16);          s += d * d;
        d = q1.y - __uint_as_float(t.z & 0xFFFF0000u);  s += d * d;
        d = q1.z - __uint_as_float(t.w << 16);          s += d * d;
        d = q1.w - __uint_as_float(t.w & 0xFFFF0000u);  s += d * d;
    }
    red[tid] = s;
    __syncthreads();
    for (int off = 128; off > 0; off >>= 1) {
        if (tid < off) red[tid] += red[tid + off];
        __syncthreads();
    }
    if (tid == 0) atomicAdd(loss, red[0] * (1.0f / 16777216.0f));
}

// ---------- GEMM3: output projection on quantized rows, upsample x2 + mask ----------
__global__ __launch_bounds__(256) void vq_out(const unsigned short* __restrict__ embed_bf,
                                              const int* __restrict__ idxbuf,
                                              const unsigned short* __restrict__ w_out_bf,
                                              const float* __restrict__ b_out,
                                              const float* __restrict__ x_mask,
                                              float* __restrict__ out) {
    __shared__ unsigned short As[2][64 * 64];
    __shared__ unsigned short Bs[2][128 * 64];
    __shared__ int codes[64];
    int tid = threadIdx.x;
    int w = tid >> 5, lane = tid & 31;
    int wm = w >> 1, wn = w & 1;
    int row0 = blockIdx.x * 64;
    if (tid < 64) codes[tid] = idxbuf[row0 + tid];
    __syncthreads();

    auto stageA = [&](int buf, int kb) {
#pragma unroll
        for (int j = 0; j < 2; ++j) {
            int chunk = tid + j * 256;
            int r = chunk >> 3, kk0 = (chunk & 7) * 8;
            cp16(embed_bf + (size_t)codes[r] * D_ + kb + kk0, &As[buf][r * 64 + kk0]);
        }
    };

    for (int nt = 0; nt < CIN_ / 128; ++nt) {
        int n0 = nt * 128;
        auto stageB = [&](int buf, int kb) {
#pragma unroll
            for (int j = 0; j < 4; ++j) {
                int chunk = tid + j * 256;
                int n = chunk >> 3, kk0 = (chunk & 7) * 8;
                cp16(w_out_bf + (size_t)(n0 + n) * D_ + kb + kk0, &Bs[buf][n * 64 + kk0]);
            }
        };
        v8f acc[4] = {};
        stageA(0, 0);
        stageB(0, 0);
        fence_all();
        int cur = 0;
        for (int kb = 0; kb < D_; kb += 64) {
            int nxt = cur ^ 1;
            bool more = (kb + 64) < D_;
            if (more) { stageA(nxt, kb + 64); stageB(nxt, kb + 64); }
            compute_tile(As[cur], Bs[cur], lane, wm, wn, acc);
            fence_all();
            cur = nxt;
        }
        int rbase = (lane >> 4) << 3;
#pragma unroll
        for (int j = 0; j < 4; ++j) {
            int oc = n0 + wn * 64 + j * 16 + (lane & 15);
            float bias = b_out[oc];
#pragma unroll
            for (int i = 0; i < 8; ++i) {
                int gr = row0 + wm * 16 + rbase + i;
                int bb = gr >> 11;          // 2048 rows per batch
                int l = gr & 2047;
                int tp = 2 * l;
                float v = acc[j][i] + bias;
                size_t ob = ((size_t)bb * CIN_ + oc) * (size_t)T_ + tp;
                float2 m = *(const float2*)(x_mask + bb * T_ + tp);
                float2 o2 = make_float2(v * m.x, v * m.y);
                *(float2*)(out + ob) = o2;   // 8B-aligned pair store (t=2l, 2l+1)
            }
        }
    }
}

extern "C" void kernel_launch(void* const* d_in, const int* in_sizes, int n_in,
                              void* d_out, int out_size, void* d_ws, size_t ws_size,
                              hipStream_t stream) {
    (void)in_sizes; (void)n_in; (void)out_size; (void)ws_size;
    const float* x      = (const float*)d_in[0];
    const float* x_mask = (const float*)d_in[1];
    const float* w_in   = (const float*)d_in[2];
    const float* b_in   = (const float*)d_in[3];
    const float* embed  = (const float*)d_in[4];
    const float* w_out  = (const float*)d_in[5];
    const float* b_out  = (const float*)d_in[6];

    float* out  = (float*)d_out;
    float* loss = out + (size_t)B_ * CIN_ * T_;   // scalar appended after main output

    char* ws = (char*)d_ws;
    unsigned short* tokens_bf = (unsigned short*)ws;                         // 32 MiB
    unsigned short* w_in_bf   = (unsigned short*)(ws + 33554432ull);         // 4 MiB
    unsigned short* embed_bf  = (unsigned short*)(ws + 37748736ull);         // 4 MiB
    unsigned short* w_out_bf  = (unsigned short*)(ws + 41943040ull);         // 2 MiB
    int*   idxbuf  = (int*)  (ws + 44040192ull);                             // 64 KiB
    float* half_e2 = (float*)(ws + 44105728ull);                             // 8 KiB

    cvt_bf16<<<1024, 256, 0, stream>>>(w_in,  w_in_bf,  (CIN_ * K1_) / 8);
    cvt_bf16<<<1024, 256, 0, stream>>>(embed, embed_bf, (NCODE * D_) / 8);
    cvt_bf16<<<512,  256, 0, stream>>>(w_out, w_out_bf, (CIN_ * D_) / 8);

    vq_prep  <<<NCODE, 256, 0, stream>>>(embed, half_e2, loss);
    vq_tokens<<<B_ * (L_ / 64) * (D_ / 128), 256, 0, stream>>>(x, w_in_bf, b_in, tokens_bf);
    vq_argmin<<<(B_ * L_) / 64, 256, 0, stream>>>(tokens_bf, embed_bf, half_e2, idxbuf);
    vq_loss  <<<1024, 256, 0, stream>>>(tokens_bf, embed, idxbuf, loss);
    vq_out   <<<(B_ * L_) / 64, 256, 0, stream>>>(embed_bf, idxbuf, w_out_bf, b_out, x_mask, out);
}
